// ConvLRUModel_29368986370550
// MI455X (gfx1250) — matrix operand order repaired and checked by reading, not compiled, not run on hardware
//
#include <hip/hip_runtime.h>
#include <cstdint>
#include <cstddef>

// ---------------------------------------------------------------------------
// ConvLRU model for MI455X (gfx1250, wave32).
//  - Complex GEMMs (DFT-as-matmul FFT2/IFFT2, channel mixes): V_WMMA_F32_16X16X4_F32
//  - Convolutions (7x7 / 3x3 / 1x1) as bf16 implicit GEMM:    V_WMMA_F32_16X16X32_BF16
// All WMMA sits in uniform control flow (EXEC all-1s requirement, ISA 7.12).
// Workspace requirement: ~337 MB (layout in kernel_launch).
// ---------------------------------------------------------------------------

typedef float  f32x2  __attribute__((ext_vector_type(2)));
typedef float  f32x8  __attribute__((ext_vector_type(8)));
typedef __bf16 bf16x8  __attribute__((ext_vector_type(8)));
typedef __bf16 bf16x16 __attribute__((ext_vector_type(16)));

static constexpr int kHW = 4096;   // 64*64 spatial plane

__device__ __forceinline__ float gelu_exact(float v) {
    return 0.5f * v * (1.0f + erff(v * 0.70710678118654752f));
}

// ---------------------------------------------------------------------------
// DFT matrix tables: D[j,k] = exp(-2*pi*i*j*k/64), Dinv = conj(D)/64
// ---------------------------------------------------------------------------
__global__ void dft_init_kernel(float* Dr, float* Di, float* Ir, float* Ii) {
    int idx = blockIdx.x * blockDim.x + threadIdx.x;
    if (idx >= 64 * 64) return;
    int j = idx >> 6, k = idx & 63;
    int ph = (j * k) & 63;                       // exact phase reduction mod 64
    float ang = -6.283185307179586f * (float)ph / 64.0f;
    float c = cosf(ang), s = sinf(ang);
    Dr[idx] = c;
    Di[idx] = s;
    Ir[idx] = c * (1.0f / 64.0f);                // conj: real unchanged
    Ii[idx] = -s * (1.0f / 64.0f);               // conj: imag negated
}

// ---------------------------------------------------------------------------
// Batched complex GEMM, K=64 fixed, one 16x16 output tile per wave.
// C = A * B (complex). A always complex.
//   BCPLX : B has imaginary part
//   WANTI : produce imaginary output plane
//   BIAS  : add per-row real bias
// Template flags keep all WMMA in uniform control flow (no saveexec regions).
// f32 WMMA fragment layouts per CDNA5 ISA 7.12.2:
//   A 16x4:  lane m=lane&15, hi=lane>>4 ; a = {A[m][k+2hi], A[m][k+2hi+1]}
//   B 4x16:  lane n=lane&15            ; b = {B[k+2hi][n],  B[k+2hi+1][n]}
//   C/D:     lane n=lane&15, vgpr r -> M = r + 8*hi
// ---------------------------------------------------------------------------
template <bool BCPLX, bool WANTI, bool BIAS>
__global__ void cgemm64_wmma_kernel(
    const float* __restrict__ Ar, const float* __restrict__ Ai, long long strideA, int lda,
    const float* __restrict__ Br, const float* __restrict__ Bi, long long strideB, int ldb,
    float* __restrict__ Cr, float* __restrict__ Ci, long long strideC, int ldc,
    const float* __restrict__ biasR,
    int M, int N, int batch)
{
    const int lane = threadIdx.x & 31;
    const int wave = threadIdx.x >> 5;
    const int mt = M >> 4, nt = N >> 4;
    long long tile  = (long long)blockIdx.x * (blockDim.x >> 5) + wave;
    long long total = (long long)batch * mt * nt;
    if (tile >= total) return;                    // wave-uniform exit
    const int b   = (int)(tile / ((long long)mt * nt));
    const int rem = (int)(tile % ((long long)mt * nt));
    const int m0  = (rem / nt) << 4;
    const int n0  = (rem % nt) << 4;

    const int mrow = lane & 15;
    const int hi   = lane >> 4;
    const int col  = n0 + mrow;

    // Hoisted base pointers; K loop steps them with constant increments.
    const float* arow_r = Ar + (long long)b * strideA + (long long)(m0 + mrow) * lda + 2 * hi;
    const float* arow_i = Ai + (long long)b * strideA + (long long)(m0 + mrow) * lda + 2 * hi;
    const float* bp_r0  = Br + (long long)b * strideB + (long long)(2 * hi) * ldb + col;
    const float* bp_r1  = bp_r0 + ldb;
    const float* bp_i0  = BCPLX ? (Bi + (long long)b * strideB + (long long)(2 * hi) * ldb + col) : nullptr;
    const float* bp_i1  = BCPLX ? (bp_i0 + ldb) : nullptr;
    const long long stepB = 4LL * ldb;

    f32x8 accR = {0.f, 0.f, 0.f, 0.f, 0.f, 0.f, 0.f, 0.f};
    f32x8 accI = accR;

    #pragma unroll 4
    for (int k = 0; k < 64; k += 4) {
        f32x2 aR = *(const f32x2*)(arow_r + k);   // contiguous K pair: one b64 load
        f32x2 aI = *(const f32x2*)(arow_i + k);

        f32x2 bR;
        bR.x = bp_r0[0];
        bR.y = bp_r1[0];

        accR = __builtin_amdgcn_wmma_f32_16x16x4_f32(false, aR, false, bR, (short)0, accR, false, false);
        if (WANTI)
            accI = __builtin_amdgcn_wmma_f32_16x16x4_f32(false, aI, false, bR, (short)0, accI, false, false);

        if (BCPLX) {
            f32x2 nI; nI.x = -aI.x; nI.y = -aI.y; // for  -Ai*Bi  term (f32 NEG bits are C-neg only)
            f32x2 bI;
            bI.x = bp_i0[0];
            bI.y = bp_i1[0];
            accR = __builtin_amdgcn_wmma_f32_16x16x4_f32(false, nI, false, bI, (short)0, accR, false, false);
            if (WANTI)
                accI = __builtin_amdgcn_wmma_f32_16x16x4_f32(false, aR, false, bI, (short)0, accI, false, false);
            bp_i0 += stepB; bp_i1 += stepB;
        }
        bp_r0 += stepB; bp_r1 += stepB;
    }

    float* cr = Cr + (long long)b * strideC;
    #pragma unroll
    for (int rr = 0; rr < 8; ++rr) {
        const int row = m0 + rr + 8 * hi;
        float v = accR[rr];
        if (BIAS) v += biasR[row];
        cr[(long long)row * ldc + col] = v;
    }
    if (WANTI) {
        float* ci = Ci + (long long)b * strideC;
        #pragma unroll
        for (int rr = 0; rr < 8; ++rr) {
            const int row = m0 + rr + 8 * hi;
            ci[(long long)row * ldc + col] = accI[rr];
        }
    }
}

// ---------------------------------------------------------------------------
// LRU recurrence along L (frequency domain):
//   h = (h + Bb) * gamma ;  h[t] = lamb * h[t-1] + h[t]
// lamb = exp(-nu)*(cos th + i sin th), per-(c,h) coefficients from params_log.
// One thread per (b, c, h, w); sequential loop over L=16. Memory-bound.
// ---------------------------------------------------------------------------
__global__ void lru_scan_kernel(float* Hr, float* Hi,
                                const float* __restrict__ pl,   // (3C, 64)
                                const float* __restrict__ Bbr,
                                const float* __restrict__ Bbi,
                                int Bsz, int Lsz, int Csz)
{
    long long idx = (long long)blockIdx.x * blockDim.x + threadIdx.x;
    long long total = (long long)Bsz * Csz * kHW;
    if (idx >= total) return;
    const int w = (int)(idx & 63);
    const int h = (int)((idx >> 6) & 63);
    const int c = (int)((idx >> 12) % Csz);
    const int b = (int)(idx / ((long long)Csz * kHW));

    const float nu = expf(pl[c * 64 + h]);
    const float th = expf(pl[(Csz + c) * 64 + h]);
    const float ga = expf(pl[(2 * Csz + c) * 64 + h]);
    const float er = expf(-nu);
    const float lr = er * cosf(th);
    const float li = er * sinf(th);
    const float br_ = Bbr[c], bi_ = Bbi[c];

    float pr = 0.f, pi = 0.f;
    for (int t = 0; t < Lsz; ++t) {
        long long off = ((((long long)b * Lsz + t) * Csz + c) * kHW) + h * 64 + w;
        float xr = (Hr[off] + br_) * ga;
        float xi = (Hi[off] + bi_) * ga;
        float sr = lr * pr - li * pi + xr;
        float si = lr * pi + li * pr + xi;
        Hr[off] = sr; Hi[off] = si;
        pr = sr; pi = si;
    }
}

// ---------------------------------------------------------------------------
// dst[plane] += LayerNorm(src[plane]) * w + b    (LN over H*W=4096, biased var)
// One 256-thread workgroup per plane.
// ---------------------------------------------------------------------------
__global__ void ln_add_kernel(const float* __restrict__ src,
                              const float* __restrict__ w,
                              const float* __restrict__ b,
                              float* dst, int planes)
{
    __shared__ float s1[256];
    __shared__ float s2[256];
    const int p = blockIdx.x;
    if (p >= planes) return;
    const float* sp = src + (long long)p * kHW;
    float sum = 0.f, ss = 0.f;
    for (int i = threadIdx.x; i < kHW; i += blockDim.x) {
        float v = sp[i]; sum += v; ss += v * v;
    }
    s1[threadIdx.x] = sum; s2[threadIdx.x] = ss;
    __syncthreads();
    for (int off = 128; off > 0; off >>= 1) {
        if ((int)threadIdx.x < off) {
            s1[threadIdx.x] += s1[threadIdx.x + off];
            s2[threadIdx.x] += s2[threadIdx.x + off];
        }
        __syncthreads();
    }
    const float mean = s1[0] * (1.0f / kHW);
    const float var  = s2[0] * (1.0f / kHW) - mean * mean;
    const float rinv = rsqrtf(var + 1e-5f);
    float* dp = dst + (long long)p * kHW;
    for (int i = threadIdx.x; i < kHW; i += blockDim.x) {
        dp[i] += (sp[i] - mean) * rinv * w[i] + b[i];
    }
}

// ---------------------------------------------------------------------------
// Activation NCHW f32 -> NHWC bf16 (conv B-fragments become contiguous loads)
// ---------------------------------------------------------------------------
__global__ void nchw_to_nhwc_bf16_kernel(const float* __restrict__ src,
                                         __bf16* __restrict__ dst,
                                         int Nf, int C)
{
    long long idx = (long long)blockIdx.x * blockDim.x + threadIdx.x;
    long long total = (long long)Nf * C * kHW;
    if (idx >= total) return;
    const int w = (int)(idx & 63);
    const int h = (int)((idx >> 6) & 63);
    const int c = (int)((idx >> 12) % C);
    const int n = (int)(idx / ((long long)C * kHW));
    dst[(((long long)n * 64 + h) * 64 + w) * C + c] = (__bf16)src[idx];
}

// Weights f32 (O, Ci, KH, KW) -> bf16 [kpos][O][Ci]
__global__ void pack_weights_bf16_kernel(const float* __restrict__ src,
                                         __bf16* __restrict__ dst,
                                         int O, int Ci, int KHW)
{
    long long idx = (long long)blockIdx.x * blockDim.x + threadIdx.x;
    long long total = (long long)O * Ci * KHW;
    if (idx >= total) return;
    const int kp = (int)(idx % KHW);
    const int ci = (int)((idx / KHW) % Ci);
    const int o  = (int)(idx / ((long long)Ci * KHW));
    dst[((long long)kp * O + o) * Ci + ci] = (__bf16)src[idx];
}

// ---------------------------------------------------------------------------
// Conv2D (SAME padding) as bf16 implicit GEMM. One wave computes a tile of
// 16 out-channels x 16 consecutive pixels (one row segment). K loop:
// kernel positions x 32-channel chunks, each a V_WMMA_F32_16X16X32_BF16.
// CIN/KH/KW compile-time -> fully unrolled channel loop, immediate offsets.
// bf16 fragment layouts used here (contiguous k order, not the ISA table; A and B match):
//   A 16x32: lane m=lane&15, hi=lane>>4: a[j]  =W[m][cb+16hi+j]  (16 contiguous ch)
//   B 32x16: lane n=lane&15            : b[j]  =X[cb+16hi+j][pixel n]  (16 contiguous ch)
// ---------------------------------------------------------------------------
template <int CIN, int KH, int KW>
__global__ void conv2d_wmma_bf16_kernel(const __bf16* __restrict__ xt,  // [Nf][64][64][CIN]
                                        const __bf16* __restrict__ wp,  // [KH*KW][Cout][CIN]
                                        const float*  __restrict__ bias,// [Cout]
                                        float* __restrict__ out,        // [Nf][Cout][64][64]
                                        int Nf, int Cout, int doGelu)
{
    const int lane = threadIdx.x & 31;
    const int wave = threadIdx.x >> 5;
    const int cot = Cout >> 4;
    const int ptiles = kHW >> 4;                 // 256 pixel tiles per frame
    long long tile  = (long long)blockIdx.x * (blockDim.x >> 5) + wave;
    long long total = (long long)Nf * cot * ptiles;
    if (tile >= total) return;                   // wave-uniform
    const int f   = (int)(tile / ((long long)cot * ptiles));
    const int rem = (int)(tile % ((long long)cot * ptiles));
    const int co0 = (rem / ptiles) << 4;
    const int pt  = rem % ptiles;
    const int h   = pt >> 2;                     // 64/16 = 4 tiles per row
    const int w0  = (pt & 3) << 4;

    const int n  = lane & 15;                    // pixel column / out-channel row
    const int hi = lane >> 4;

    f32x8 acc = {0.f, 0.f, 0.f, 0.f, 0.f, 0.f, 0.f, 0.f};

    const __bf16* wrow  = wp + ((long long)co0 + n) * CIN;            // + kp*Cout*CIN
    const __bf16* xbase = xt + ((long long)f * 64) * 64 * CIN;        // frame base

    constexpr int py = KH >> 1, px = KW >> 1;
    for (int kp = 0; kp < KH * KW; ++kp) {
        const int dy = kp / KW - py;
        const int dx = kp % KW - px;
        const int ih = h + dy;
        if (ih < 0 || ih >= 64) continue;        // whole tile shares row h (uniform)
        const int iw = w0 + n + dx;
        const bool inb = (iw >= 0 && iw < 64);
        const int iwc = inb ? iw : 0;            // clamped address, masked after load
        const __bf16* awp = wrow + (long long)kp * Cout * CIN;
        const __bf16* bpp = xbase + ((long long)ih * 64 + iwc) * CIN + 16 * hi;
        #pragma unroll
        for (int cb = 0; cb < CIN; cb += 32) {
            // A fragment: weight row o = co0+n
            bf16x16 a = *(const bf16x16*)(awp + cb + 16 * hi);
            // B fragment: 16 contiguous channels at this lane's pixel
            bf16x16 bfr = *(const bf16x16*)(bpp + cb);
            if (!inb) {
                #pragma unroll
                for (int j = 0; j < 16; ++j) bfr[j] = (__bf16)0.0f;   // zero padding
            }
            acc = __builtin_amdgcn_wmma_f32_16x16x32_bf16(false, a, false, bfr, (short)0, acc, false, false);
        }
    }

    #pragma unroll
    for (int rr = 0; rr < 8; ++rr) {
        const int o = co0 + rr + 8 * hi;
        float v = acc[rr] + bias[o];
        if (doGelu) v = gelu_exact(v);
        out[(((long long)f * Cout + o) * 64 + h) * 64 + w0 + n] = v;
    }
}

// ---------------------------------------------------------------------------
// Host orchestration
// ---------------------------------------------------------------------------
static inline unsigned cdiv(long long a, long long b) { return (unsigned)((a + b - 1) / b); }

extern "C" void kernel_launch(void* const* d_in, const int* in_sizes, int n_in,
                              void* d_out, int out_size, void* d_ws, size_t ws_size,
                              hipStream_t stream)
{
    (void)in_sizes; (void)n_in; (void)out_size; (void)ws_size;
    const int Bsz = 4, Lsz = 16, Csz = 64, Fsz = 128, NHh = 2, NBb = 2;
    const long long NEL = (long long)Bsz * Lsz * Csz * kHW;   // 16,777,216 elements

    const float* in_x       = (const float*)d_in[0];
    const float* params_log = (const float*)d_in[1];
    const float* pBw_r = (const float*)d_in[2];
    const float* pBw_i = (const float*)d_in[3];
    const float* pBb_r = (const float*)d_in[4];
    const float* pBb_i = (const float*)d_in[5];
    const float* pCw_r = (const float*)d_in[6];
    const float* pCw_i = (const float*)d_in[7];
    const float* pCb_r = (const float*)d_in[8];
    const float* pCb_i = (const float*)d_in[9];
    const float* lru_ln_w = (const float*)d_in[10];
    const float* lru_ln_b = (const float*)d_in[11];
    const float* cin_w  = (const float*)d_in[12];
    const float* cin_b  = (const float*)d_in[13];
    const float* h3_w   = (const float*)d_in[14];
    const float* h3_b   = (const float*)d_in[15];
    const float* h1_w   = (const float*)d_in[16];
    const float* h1_b   = (const float*)d_in[17];
    const float* h_ln_w = (const float*)d_in[18];
    const float* h_ln_b = (const float*)d_in[19];
    const float* cout_w = (const float*)d_in[20];
    const float* cout_b = (const float*)d_in[21];
    const float* ffn_ln_w = (const float*)d_in[22];
    const float* ffn_ln_b = (const float*)d_in[23];

    float* X = (float*)d_out;                    // running activation (B,L,C,H,W)

    // ---- workspace layout (~337 MB) ----
    char* ws = (char*)d_ws;
    float* R0r = (float*)ws;                     // complex ping buf / FFN y_ (f32, 2*NEL)
    float* R0i = R0r + NEL;
    float* R1r = R0i + NEL;                      // complex pong buf / FFN t  (f32, 2*NEL)
    float* R1i = R1r + NEL;
    float* R2f = R1i + NEL;                      // Cw-mix real out (NEL f32) / NHWC bf16 stage
    __bf16* XT = (__bf16*)R2f;                   // 2*NEL bf16 (same bytes)
    float* Dr = (float*)(ws + (size_t)5 * NEL * sizeof(float));
    float* Di = Dr + 4096;
    float* Ir = Di + 4096;
    float* Ii = Ir + 4096;
    __bf16* WP7 = (__bf16*)(Ii + 4096);          // 49*128*64
    __bf16* WP3 = WP7 + 49 * 128 * 64;           // 9*128*128
    __bf16* WP1 = WP3 + 9 * 128 * 128;           // 128*128
    __bf16* WPC = WP1 + 128 * 128;               // 64*128

    // x -> d_out (running buffer)
    hipMemcpyAsync(X, in_x, (size_t)NEL * sizeof(float), hipMemcpyDeviceToDevice, stream);
    dft_init_kernel<<<16, 256, 0, stream>>>(Dr, Di, Ir, Ii);

    const long long slices = (long long)Bsz * Lsz * Csz;       // 4096 FFT slices
    const long long fftTiles = slices * 4 * 4;                 // 16x16 tiles, M=N=64
    const long long mixTiles = (long long)Bsz * Lsz * 4 * 256; // M=64, N=4096, batch=64
    const long long scanThreads = (long long)Bsz * Csz * kHW;

    for (int i = 0; i < NBb; ++i) {
        // ===================== ConvLRU layer =====================
        // FFT2 stage 1: T = D * X  (B real, complex out)
        cgemm64_wmma_kernel<false, true, false><<<cdiv(fftTiles, 8), 256, 0, stream>>>(
            Dr, Di, 0, 64,  X, nullptr, kHW, 64,
            R0r, R0i, kHW, 64,  nullptr, 64, 64, (int)slices);
        // FFT2 stage 2: H = T * D
        cgemm64_wmma_kernel<true, true, false><<<cdiv(fftTiles, 8), 256, 0, stream>>>(
            R0r, R0i, kHW, 64,  Dr, Di, 0, 64,
            R1r, R1i, kHW, 64,  nullptr, 64, 64, (int)slices);
        // Channel mix: H = Bw * H  per (b,l): (64 x 4096)
        cgemm64_wmma_kernel<true, true, false><<<cdiv(mixTiles, 8), 256, 0, stream>>>(
            pBw_r + (long long)i * 4096, pBw_i + (long long)i * 4096, 0, 64,
            R1r, R1i, (long long)Csz * kHW, kHW,
            R0r, R0i, (long long)Csz * kHW, kHW,
            nullptr, 64, kHW, Bsz * Lsz);
        // Bias + gamma + recurrence along L (in place)
        lru_scan_kernel<<<cdiv(scanThreads, 256), 256, 0, stream>>>(
            R0r, R0i, params_log + (long long)i * 3 * 64 * 64,
            pBb_r + (long long)i * 64, pBb_i + (long long)i * 64, Bsz, Lsz, Csz);
        // IFFT2 stage 1: T = Dinv * H
        cgemm64_wmma_kernel<true, true, false><<<cdiv(fftTiles, 8), 256, 0, stream>>>(
            Ir, Ii, 0, 64,  R0r, R0i, kHW, 64,
            R1r, R1i, kHW, 64,  nullptr, 64, 64, (int)slices);
        // IFFT2 stage 2: Y = T * Dinv
        cgemm64_wmma_kernel<true, true, false><<<cdiv(fftTiles, 8), 256, 0, stream>>>(
            R1r, R1i, kHW, 64,  Ir, Ii, 0, 64,
            R0r, R0i, kHW, 64,  nullptr, 64, 64, (int)slices);
        // Channel mix: real(Cw * Y) + Cb_r  (real-only -> half the WMMAs)
        cgemm64_wmma_kernel<true, false, true><<<cdiv(mixTiles, 8), 256, 0, stream>>>(
            pCw_r + (long long)i * 4096, pCw_i + (long long)i * 4096, 0, 64,
            R0r, R0i, (long long)Csz * kHW, kHW,
            R2f, nullptr, (long long)Csz * kHW, kHW,
            pCb_r + (long long)i * 64, 64, kHW, Bsz * Lsz);
        // x = LN(real) + x
        ln_add_kernel<<<(unsigned)slices, 256, 0, stream>>>(
            R2f, lru_ln_w + (long long)i * kHW, lru_ln_b + (long long)i * kHW, X, (int)slices);

        // ===================== FeedForward =====================
        const int Nf = Bsz * Lsz;                 // 64 frames
        // conv-in 7x7: C->F, gelu  :: y_ = R0
        pack_weights_bf16_kernel<<<cdiv((long long)Fsz * Csz * 49, 256), 256, 0, stream>>>(
            cin_w + (long long)i * Fsz * Csz * 49, WP7, Fsz, Csz, 49);
        nchw_to_nhwc_bf16_kernel<<<cdiv((long long)Nf * Csz * kHW, 256), 256, 0, stream>>>(X, XT, Nf, Csz);
        conv2d_wmma_bf16_kernel<64, 7, 7><<<cdiv((long long)Nf * (Fsz / 16) * 256, 8), 256, 0, stream>>>(
            XT, WP7, cin_b + (long long)i * Fsz, R0r, Nf, Fsz, 1);

        for (int j = 0; j < NHh; ++j) {
            const long long wj = (long long)(i * NHh + j);
            pack_weights_bf16_kernel<<<cdiv((long long)Fsz * Fsz * 9, 256), 256, 0, stream>>>(
                h3_w + wj * Fsz * Fsz * 9, WP3, Fsz, Fsz, 9);
            pack_weights_bf16_kernel<<<cdiv((long long)Fsz * Fsz, 256), 256, 0, stream>>>(
                h1_w + wj * Fsz * Fsz, WP1, Fsz, Fsz, 1);
            // t = gelu(conv3(y_))
            nchw_to_nhwc_bf16_kernel<<<cdiv((long long)Nf * Fsz * kHW, 256), 256, 0, stream>>>(R0r, XT, Nf, Fsz);
            conv2d_wmma_bf16_kernel<128, 3, 3><<<cdiv((long long)Nf * (Fsz / 16) * 256, 8), 256, 0, stream>>>(
                XT, WP3, h3_b + wj * Fsz, R1r, Nf, Fsz, 1);
            // t = gelu(1x1(t))
            nchw_to_nhwc_bf16_kernel<<<cdiv((long long)Nf * Fsz * kHW, 256), 256, 0, stream>>>(R1r, XT, Nf, Fsz);
            conv2d_wmma_bf16_kernel<128, 1, 1><<<cdiv((long long)Nf * (Fsz / 16) * 256, 8), 256, 0, stream>>>(
                XT, WP1, h1_b + wj * Fsz, R1r, Nf, Fsz, 1);
            // y_ += LN(t)
            ln_add_kernel<<<(unsigned)(Nf * Fsz), 256, 0, stream>>>(
                R1r, h_ln_w + wj * kHW, h_ln_b + wj * kHW, R0r, Nf * Fsz);
        }
        // cout 1x1: F->C (no gelu), then x += LN(.)
        pack_weights_bf16_kernel<<<cdiv((long long)Csz * Fsz, 256), 256, 0, stream>>>(
            cout_w + (long long)i * Csz * Fsz, WPC, Csz, Fsz, 1);
        nchw_to_nhwc_bf16_kernel<<<cdiv((long long)Nf * Fsz * kHW, 256), 256, 0, stream>>>(R0r, XT, Nf, Fsz);
        conv2d_wmma_bf16_kernel<128, 1, 1><<<cdiv((long long)Nf * (Csz / 16) * 256, 8), 256, 0, stream>>>(
            XT, WPC, cout_b + (long long)i * Csz, R1r, Nf, Csz, 0);
        ln_add_kernel<<<(unsigned)slices, 256, 0, stream>>>(
            R1r, ffn_ln_w + (long long)i * kHW, ffn_ln_b + (long long)i * kHW, X, (int)slices);
    }
}
